// FP8Linear_68607807586976
// MI455X (gfx1250) — matrix operand order repaired and checked by reading, not compiled, not run on hardware
//
#include <hip/hip_runtime.h>

// Problem dims (fixed by the reference).
#define M_DIM 8192
#define N_DIM 4096
#define K_DIM 4096

// Block tile: 128(M) x 256(N), 8 waves (2x4), wave tile 64x64 = 4x4 WMMA accums.
#define KSTEP 32
#define A_ROWS 128
#define B_ROWS 256
#define LDS_PITCH 40            // 32 K-elements + 8 pad (80B rows -> conflict-free ds_load)

typedef __attribute__((ext_vector_type(16))) __bf16 v16bf;
typedef __attribute__((ext_vector_type(8)))  __bf16 v8bf;
typedef __attribute__((ext_vector_type(4)))  __bf16 v4bf;
typedef __attribute__((ext_vector_type(8)))  float  v8f;
typedef __attribute__((ext_vector_type(4)))  float  v4f;

// ---------------------------------------------------------------------------
// Pass 1: quantize f32 master weights through fp8 e4m3 (RNE, saturating) and
// store the dequantized value as bf16 (exact: e4m3 values are a subset of
// bf16). Memory-bound: ~96 MB traffic ~= 4 us at 23.3 TB/s.
// ---------------------------------------------------------------------------
__device__ __forceinline__ float quant_e4m3_roundtrip(float x) {
    unsigned u    = __float_as_uint(x);
    unsigned sign = u & 0x80000000u;
    unsigned mag  = u & 0x7FFFFFFFu;
    if (mag == 0u) return x;                                   // +-0
    if (mag >= 0x7F800000u)                                    // inf/NaN -> NaN (e4m3fn)
        return __uint_as_float(sign | 0x7FC00000u);
    int exp = (int)(mag >> 23) - 127;
    if (exp < -10)                                             // underflow to zero
        return __uint_as_float(sign);
    if (exp == -10) {                                          // round to 0 or 2^-9
        return (mag > 0x3A800000u) ? __uint_as_float(sign | 0x3B000000u)
                                   : __uint_as_float(sign);
    }
    int shift = (exp >= -6) ? 20 : (20 + (-6 - exp));          // bits to drop (<=23)
    unsigned half = 1u << (shift - 1);
    unsigned r = mag + (half - 1u) + ((mag >> shift) & 1u);    // round-to-nearest-even
    r &= ~((1u << shift) - 1u);
    if (r > 0x43E00000u) r = 0x43E00000u;                      // saturate at 448.0
    return __uint_as_float(sign | r);
}

__global__ __launch_bounds__(256)
void quantize_weights_kernel(const float* __restrict__ W, __bf16* __restrict__ Wq, int n) {
    int i = (blockIdx.x * blockDim.x + threadIdx.x) * 4;
    if (i + 3 >= n) {
        for (int j = i; j < n; ++j) Wq[j] = (__bf16)quant_e4m3_roundtrip(W[j]);
        return;
    }
    v4f v = *(const v4f*)(W + i);
    v4bf o;
    o.x = (__bf16)quant_e4m3_roundtrip(v.x);
    o.y = (__bf16)quant_e4m3_roundtrip(v.y);
    o.z = (__bf16)quant_e4m3_roundtrip(v.z);
    o.w = (__bf16)quant_e4m3_roundtrip(v.w);
    *(v4bf*)(Wq + i) = o;
}

// ---------------------------------------------------------------------------
// Async global->LDS panel fill (CDNA5 async-copy path, ASYNCcnt-tracked).
// Panels stored [row][32 elems] with 40-element pitch. Each of 256 threads
// moves 2x16B of A and 4x16B of B per K-step.
// ---------------------------------------------------------------------------
__device__ __forceinline__ void async_fill_panels(
    const __bf16* __restrict__ X, const __bf16* __restrict__ Wq,
    int blockM0, int blockN0, int k0,
    unsigned ldsA, unsigned ldsB, int tid)
{
#pragma unroll
    for (int j = 0; j < 2; ++j) {                 // A: 512 chunks of 16B
        int chunk = tid + j * 256;
        int row   = chunk >> 2;                   // 0..127
        int off   = (chunk & 3) * 16;             // byte offset within 64B row
        unsigned long long g = (unsigned long long)(size_t)
            ((const char*)(X + (size_t)(blockM0 + row) * K_DIM + k0) + off);
        unsigned l = ldsA + (unsigned)(row * (LDS_PITCH * 2) + off);
        asm volatile("global_load_async_to_lds_b128 %0, %1, off"
                     :: "v"(l), "v"(g) : "memory");
    }
#pragma unroll
    for (int j = 0; j < 4; ++j) {                 // B: 1024 chunks of 16B
        int chunk = tid + j * 256;
        int row   = chunk >> 2;                   // 0..255
        int off   = (chunk & 3) * 16;
        unsigned long long g = (unsigned long long)(size_t)
            ((const char*)(Wq + (size_t)(blockN0 + row) * K_DIM + k0) + off);
        unsigned l = ldsB + (unsigned)(row * (LDS_PITCH * 2) + off);
        asm volatile("global_load_async_to_lds_b128 %0, %1, off"
                     :: "v"(l), "v"(g) : "memory");
    }
}

// ---------------------------------------------------------------------------
// Pass 2: bf16 WMMA GEMM with double-buffered async LDS staging. The K-loop
// is unrolled 2x so each phase reads a compile-time-known LDS buffer
// (guarantees ds_load_b128, not flat_load, for fragment gathers).
// ---------------------------------------------------------------------------
__global__ __launch_bounds__(256, 1)
void fp8linear_wmma_kernel(const __bf16* __restrict__ X,   // [M, K] bf16
                           const __bf16* __restrict__ Wq,  // [N, K] bf16 (quantized)
                           const float*  __restrict__ scale,
                           __bf16* __restrict__ Y)         // [M, N] bf16
{
    __shared__ __align__(16) __bf16 ldsA[2][A_ROWS * LDS_PITCH];  // 2 x 10 KB
    __shared__ __align__(16) __bf16 ldsB[2][B_ROWS * LDS_PITCH];  // 2 x 20 KB

    const int tid      = threadIdx.x;
    const int lane     = tid & 31;
    const int wave     = tid >> 5;
    const int laneIdx  = lane & 15;      // row-in-tile (A) / col-in-tile (B,C)
    const int laneHalf = lane >> 4;      // K-chunk select (A,B) / M-half (C)

    const int waveM   = wave >> 2;       // 0..1
    const int waveN   = wave & 3;        // 0..3
    const int blockM0 = blockIdx.y * 128;
    const int blockN0 = blockIdx.x * 256;

    // Low 32 bits of a generic LDS pointer are the LDS byte address (for the
    // async-copy VDST operand).
    const unsigned lA0 = (unsigned)(size_t)&ldsA[0][0];
    const unsigned lA1 = (unsigned)(size_t)&ldsA[1][0];
    const unsigned lB0 = (unsigned)(size_t)&ldsB[0][0];
    const unsigned lB1 = (unsigned)(size_t)&ldsB[1][0];

    // A(16x32) layout matching B: lanes 0-15 hold K {0..15}, lanes 16-31
    // hold {16..31} -> two 16B LDS reads at +laneHalf*16 (+0/+8 elems).
    const int aOff = (waveM * 64 + laneIdx) * LDS_PITCH + laneHalf * 16;
    // B(32x16) layout matching A (not ISA order): lane holds 16 contiguous K of column
    // laneIdx; lanes 16-31 hold K+16 -> one 32B LDS read at +laneHalf*16.
    const int bOff = (waveN * 64 + laneIdx) * LDS_PITCH + laneHalf * 16;

    v8f acc[4][4] = {};

    // One WMMA K-step from a compile-time-selected LDS buffer.
    auto compute_step = [&](const __bf16* aPanel, const __bf16* bPanel) {
        v16bf afrag[4], bfrag[4];
#pragma unroll
        for (int mt = 0; mt < 4; ++mt) {
            const __bf16* p = aPanel + aOff + mt * 16 * LDS_PITCH;
            v8bf lo = *(const v8bf*)(p);
            v8bf hi = *(const v8bf*)(p + 8);
            afrag[mt] = __builtin_shufflevector(lo, hi,
                0, 1, 2, 3, 4, 5, 6, 7, 8, 9, 10, 11, 12, 13, 14, 15);
        }
#pragma unroll
        for (int nt = 0; nt < 4; ++nt) {
            bfrag[nt] = *(const v16bf*)(bPanel + bOff + nt * 16 * LDS_PITCH);
        }
#pragma unroll
        for (int mt = 0; mt < 4; ++mt) {
#pragma unroll
            for (int nt = 0; nt < 4; ++nt) {
                acc[mt][nt] = __builtin_amdgcn_wmma_f32_16x16x32_bf16(
                    /*neg_a=*/false, afrag[mt],
                    /*neg_b=*/false, bfrag[nt],
                    /*c_mod=*/(short)0, acc[mt][nt],
                    /*reuse_a=*/false, /*reuse_b=*/false);
            }
        }
    };

    // Prologue: fill panel 0 into buffer 0.
    async_fill_panels(X, Wq, blockM0, blockN0, 0, lA0, lB0, tid);

    const int ksteps = K_DIM / KSTEP;    // 128 (even)
    for (int k = 0; k < ksteps; k += 2) {
        // ---- phase 0: consume buffer 0, prefetch k+1 into buffer 1 ----
        asm volatile("s_wait_asynccnt 0x0" ::: "memory");
        __syncthreads();
        async_fill_panels(X, Wq, blockM0, blockN0, (k + 1) * KSTEP, lA1, lB1, tid);
        compute_step(&ldsA[0][0], &ldsB[0][0]);

        // ---- phase 1: consume buffer 1, prefetch k+2 into buffer 0 ----
        asm volatile("s_wait_asynccnt 0x0" ::: "memory");
        __syncthreads();
        if (k + 2 < ksteps)
            async_fill_panels(X, Wq, blockM0, blockN0, (k + 2) * KSTEP, lA0, lB0, tid);
        compute_step(&ldsA[1][0], &ldsB[1][0]);
    }

    const float s = scale[0];

    // C/D layout: VGPR r, lane l -> M = r + 8*(l>>4), N = l&15 (within tile).
#pragma unroll
    for (int mt = 0; mt < 4; ++mt) {
#pragma unroll
        for (int nt = 0; nt < 4; ++nt) {
            const int row0 = blockM0 + waveM * 64 + mt * 16 + laneHalf * 8;
            const int col  = blockN0 + waveN * 64 + nt * 16 + laneIdx;
#pragma unroll
            for (int r = 0; r < 8; ++r) {
                Y[(size_t)(row0 + r) * N_DIM + col] = (__bf16)(acc[mt][nt][r] * s);
            }
        }
    }
}

// ---------------------------------------------------------------------------
// Harness entry point.
//   d_in[0]: x      bf16 [M*K]
//   d_in[1]: weight f32  [N*K]
//   d_in[2]: scale  f32  [1]
//   d_out  : y      bf16 [M*N]
//   d_ws   : scratch for bf16-dequantized e4m3 weights (N*K*2 = 32 MB)
// ---------------------------------------------------------------------------
extern "C" void kernel_launch(void* const* d_in, const int* in_sizes, int n_in,
                              void* d_out, int out_size, void* d_ws, size_t ws_size,
                              hipStream_t stream) {
    const __bf16* X     = (const __bf16*)d_in[0];
    const float*  W     = (const float*)d_in[1];
    const float*  scale = (const float*)d_in[2];
    __bf16*       Y     = (__bf16*)d_out;
    __bf16*       Wq    = (__bf16*)d_ws;

    const int nW = in_sizes[1];  // N*K
    const int qThreads = 256;
    const int qBlocks  = (nW / 4 + qThreads - 1) / qThreads;
    quantize_weights_kernel<<<qBlocks, qThreads, 0, stream>>>(W, Wq, nW);

    dim3 grid(N_DIM / 256, M_DIM / 128);   // (16, 64)
    fp8linear_wmma_kernel<<<grid, 256, 0, stream>>>(X, Wq, scale, Y);
}
